// fuGenerator_5334349382358
// MI455X (gfx1250) — matrix operand order repaired and checked by reading, not compiled, not run on hardware
//
#include <hip/hip_runtime.h>
#include <hip/hip_bf16.h>
#include <math.h>

typedef __attribute__((ext_vector_type(16))) _Float16 v16h;
typedef __attribute__((ext_vector_type(8)))  _Float16 v8h;
typedef __attribute__((ext_vector_type(8)))  float    v8f;

#define WS_    7
#define SHIFT_ 3
#define NH     6
#define HD     32
#define C_     192
#define N_     49
#define B_IMG  32
#define HW     56
#define NWIN   64           // windows per image (8x8)
#define NWTOT  2048         // B_IMG * NWIN
#define MROWS  100352       // NWTOT * N_
#define SCALE_Q 0.17677669529663687f

// Use inline asm for the CDNA5 async global->LDS path (builtin signature is
// toolchain-specific; asm is portable across both toolchains).
#define USE_ASYNC_LDS_ASM 1

static __device__ inline v8f vzero8() {
    v8f z;
#pragma unroll
    for (int i = 0; i < 8; ++i) z[i] = 0.0f;
    return z;
}

static __device__ inline v8f wmma16x16x32(v16h a, v16h b, v8f c) {
    return __builtin_amdgcn_wmma_f32_16x16x32_f16(false, a, false, b, (short)0, c,
                                                  false, false);
}

// ---------------------------------------------------------------------------
// CDNA5 async global->LDS copy (16 bytes / lane), ASYNCcnt-tracked.
// LDS byte address = low 32 bits of the generic pointer (ISA aperture rule).
// ---------------------------------------------------------------------------
static __device__ inline void async_copy16(const _Float16* g, _Float16* l) {
#if USE_ASYNC_LDS_ASM
    unsigned loff = (unsigned)(unsigned long long)l;
    asm volatile("global_load_async_to_lds_b128 %0, %1, off"
                 :: "v"(loff), "v"(g)
                 : "memory");
#else
    *(v8h*)l = *(const v8h*)g;
#endif
}

static __device__ inline void async_wait_all() {
#if USE_ASYNC_LDS_ASM
    asm volatile("s_wait_asynccnt 0x0" ::: "memory");
#endif
}

// ---------------------------------------------------------------------------
// Kernel 0: convert + transpose weights to f16 [K, Nout] row-major
// ---------------------------------------------------------------------------
__global__ __launch_bounds__(256) void prep_weights(
    const float* __restrict__ qkv_w,   // [576,192]
    const float* __restrict__ proj_w,  // [192,192]
    const float* __restrict__ fc1_w,   // [192,192]
    _Float16* __restrict__ wqT,        // [192,576]
    _Float16* __restrict__ wpT,        // [192,192]
    _Float16* __restrict__ wfT)        // [192,192]
{
    int idx = blockIdx.x * 256 + threadIdx.x;
    const int NQ = 576 * 192, NP = 192 * 192;
    if (idx < NQ) {
        int n = idx / 192, k = idx % 192;
        wqT[k * 576 + n] = (_Float16)qkv_w[idx];
    } else if (idx < NQ + NP) {
        int j = idx - NQ;
        int n = j / 192, k = j % 192;
        wpT[k * 192 + n] = (_Float16)proj_w[j];
    } else if (idx < NQ + 2 * NP) {
        int j = idx - NQ - NP;
        int n = j / 192, k = j % 192;
        wfT[k * 192 + n] = (_Float16)fc1_w[j];
    }
}

// ---------------------------------------------------------------------------
// Kernel 1: LayerNorm + cyclic shift(-3,-3) + window partition -> xw f16
// ---------------------------------------------------------------------------
__global__ __launch_bounds__(256) void ln_shift_partition(
    const float* __restrict__ x,   // [B, 3136, 192]
    const float* __restrict__ g,
    const float* __restrict__ b,
    _Float16* __restrict__ xw)     // [MROWS, 192]
{
    int wid = threadIdx.x >> 5, lane = threadIdx.x & 31;
    int s = blockIdx.x * 8 + wid;               // window-order row
    int win = s / N_, n = s % N_;
    int img = win / NWIN, wi = win % NWIN;
    int wh = wi >> 3, ww = wi & 7;
    int r = n / WS_, c = n % WS_;
    int sh = wh * WS_ + r, sw = ww * WS_ + c;   // shifted-image coords
    int hs = (sh + SHIFT_) % HW, ws2 = (sw + SHIFT_) % HW;
    size_t src = (size_t)img * (HW * HW) + (size_t)hs * HW + ws2;

    const float* xr = x + src * C_;
    float v[6], sum = 0.f, sq = 0.f;
#pragma unroll
    for (int i = 0; i < 6; ++i) {
        v[i] = xr[lane + 32 * i];
        sum += v[i];
        sq  += v[i] * v[i];
    }
#pragma unroll
    for (int m = 1; m < 32; m <<= 1) {
        sum += __shfl_xor(sum, m, 32);
        sq  += __shfl_xor(sq,  m, 32);
    }
    float mean = sum * (1.0f / C_);
    float var  = sq * (1.0f / C_) - mean * mean;
    float rstd = rsqrtf(var + 1e-5f);
    _Float16* orow = xw + (size_t)s * C_;
#pragma unroll
    for (int i = 0; i < 6; ++i) {
        int ch = lane + 32 * i;
        orow[ch] = (_Float16)((v[i] - mean) * rstd * g[ch] + b[ch]);
    }
}

// ---------------------------------------------------------------------------
// Kernel 2: WMMA GEMM  out[M,Nout] = A[M,192] @ Bt[192,Nout] (+ epilogue)
// block = 256 thr (8 waves); tile 128 x 64.
// B panel (192x64 f16 = 24KB) staged once in LDS via async-to-LDS;
// A fragments register-double-buffered across the 6 K-steps.
// MODE 0: QKV   (+bias, scale q cols<192, f16 out [M,576])
// MODE 1: PROJ  (+bias, reverse-shift scatter to [B,L,C] f16)
// MODE 2: FC1   (+bias, exact GELU, + shortcut f32, f32 out)
// ---------------------------------------------------------------------------
template <int MODE>
__global__ __launch_bounds__(256) void gemm_wmma(
    const _Float16* __restrict__ A,
    const _Float16* __restrict__ Bt,
    const float* __restrict__ bias,
    int Nout,
    const float* __restrict__ shortcut,
    float* __restrict__ outF,
    _Float16* __restrict__ outH)
{
    const int K = C_;
    __shared__ __align__(16) _Float16 Bp[192][64];   // 24 KB

    int tid = threadIdx.x;
    int wid = tid >> 5, lane = tid & 31;
    int mbase = blockIdx.x * 128 + wid * 16;
    int nb = blockIdx.y * 64;
    int lg = lane >> 4, ncol = lane & 15;

    // ---- stage whole B panel to LDS (async, 16B per op) ----
#pragma unroll
    for (int i = 0; i < 6; ++i) {
        int idx = i * 256 + tid;          // 0..1535 chunks of 8 halves
        int row = idx >> 3;               // 0..191
        int seg = (idx & 7) * 8;          // 0,8,...,56
        async_copy16(Bt + (size_t)row * Nout + nb + seg, &Bp[row][seg]);
    }
    async_wait_all();
    __syncthreads();

    v8f acc[4];
#pragma unroll
    for (int t = 0; t < 4; ++t) acc[t] = vzero8();

    // ---- A fragment double buffer ----
    const _Float16* arow = A + (size_t)(mbase + ncol) * K + lg * 8;
    v8h alo = *(const v8h*)(arow);
    v8h ahi = *(const v8h*)(arow + 16);

#pragma unroll
    for (int kb = 0; kb < K; kb += 32) {
        v8h nlo = alo, nhi = ahi;
        if (kb + 32 < K) {                     // prefetch next K-step
            nlo = *(const v8h*)(arow + kb + 32);
            nhi = *(const v8h*)(arow + kb + 48);
        }
        v16h a;
#pragma unroll
        for (int i = 0; i < 8; ++i) { a[i] = alo[i]; a[i + 8] = ahi[i]; }

#pragma unroll
        for (int t = 0; t < 4; ++t) {
            v16h bb;
#pragma unroll
            for (int i = 0; i < 8; ++i) {
                bb[i] = Bp[kb + lg * 8 + i][t * 16 + ncol];
                bb[i + 8] = Bp[kb + 16 + lg * 8 + i][t * 16 + ncol];
            }
            acc[t] = wmma16x16x32(a, bb, acc[t]);
        }
        alo = nlo; ahi = nhi;
    }

    // ---- epilogue ----
#pragma unroll
    for (int t = 0; t < 4; ++t) {
        int n = nb + t * 16 + ncol;
        float bn = bias[n];
#pragma unroll
        for (int r = 0; r < 8; ++r) {
            int m = mbase + lg * 8 + r;
            float val = acc[t][r] + bn;
            if (MODE == 0) {                       // QKV
                if (n < C_) val *= SCALE_Q;        // scale q
                outH[(size_t)m * 576 + n] = (_Float16)val;
            } else if (MODE == 1) {                // PROJ + reverse shift
                int win = m / N_, p = m % N_;
                int img = win / NWIN, wi = win % NWIN;
                int wh = wi >> 3, ww = wi & 7;
                int rr = p / WS_, cc = p % WS_;
                int sh = wh * WS_ + rr, sw = ww * WS_ + cc;
                int hh = (sh + SHIFT_) % HW, w2 = (sw + SHIFT_) % HW;
                size_t dst = (size_t)img * (HW * HW) + (size_t)hh * HW + w2;
                outH[dst * C_ + n] = (_Float16)val;
            } else {                               // FC1 + GELU + residual
                float gl = 0.5f * val * (1.0f + erff(val * 0.70710678118654752f));
                size_t o = (size_t)m * C_ + n;
                outF[o] = shortcut[o] + gl;
            }
        }
    }
}

// ---------------------------------------------------------------------------
// Kernel 3: windowed attention, one block per (window, head); 4 waves
// ---------------------------------------------------------------------------
__global__ __launch_bounds__(128) void attn_kernel(
    const _Float16* __restrict__ qkv,   // [MROWS, 576]
    const float* __restrict__ rpb,      // [169, 6]
    const float* __restrict__ amask,    // [64, 49, 49]
    _Float16* __restrict__ out)         // [MROWS, 192]
{
    __shared__ __align__(16) _Float16 Klds[64][32];
    __shared__ __align__(16) _Float16 Vlds[64][32];
    __shared__ _Float16 Plds[64][72];

    int win = blockIdx.x, head = blockIdx.y;
    int t = threadIdx.x;
    int wid = t >> 5, lane = t & 31;
    int lg = lane >> 4, ncol = lane & 15;

    // ---- stage K, V in LDS via async copies (zero-pad rows >= 49) ----
    {
        int row = t >> 1, ch = (t & 1) * 16;
        if (row < N_) {
            const _Float16* base =
                qkv + (size_t)(win * N_ + row) * 576 + head * HD + ch;
            async_copy16(base + C_,         &Klds[row][ch]);
            async_copy16(base + C_ + 8,     &Klds[row][ch + 8]);
            async_copy16(base + 2 * C_,     &Vlds[row][ch]);
            async_copy16(base + 2 * C_ + 8, &Vlds[row][ch + 8]);
        } else {
            v8h z;
#pragma unroll
            for (int i = 0; i < 8; ++i) z[i] = (_Float16)0.0f;
            *(v8h*)&Klds[row][ch] = z; *(v8h*)&Klds[row][ch + 8] = z;
            *(v8h*)&Vlds[row][ch] = z; *(v8h*)&Vlds[row][ch + 8] = z;
        }
    }
    async_wait_all();
    __syncthreads();

    int mtile = wid * 16;

    // ---- q A-fragment direct from global (q already scaled) ----
    int mq = mtile + ncol; if (mq > N_ - 1) mq = N_ - 1;
    const _Float16* qr =
        qkv + (size_t)(win * N_ + mq) * 576 + head * HD + lg * 8;
    v16h aq;
    {
        v8h qlo = *(const v8h*)qr;
        v8h qhi = *(const v8h*)(qr + 16);
#pragma unroll
        for (int i = 0; i < 8; ++i) { aq[i] = qlo[i]; aq[i + 8] = qhi[i]; }
    }

    // ---- scores: 4 tiles of 16 cols, K=32 in one WMMA each ----
    v8f s[4];
#pragma unroll
    for (int nt = 0; nt < 4; ++nt) {
        v16h bk;
#pragma unroll
        for (int j = 0; j < 16; ++j) bk[j] = Klds[nt * 16 + ncol][(j < 8 ? 0 : 16) + lg * 8 + (j & 7)];
        s[nt] = wmma16x16x32(aq, bk, vzero8());
    }

    // ---- bias + mask + padding ----
#pragma unroll
    for (int nt = 0; nt < 4; ++nt) {
#pragma unroll
        for (int r = 0; r < 8; ++r) {
            int m = mtile + lg * 8 + r;
            int n = nt * 16 + ncol;
            float v;
            if (m < N_ && n < N_) {
                int rm = m / WS_, cm = m % WS_, rn = n / WS_, cn = n % WS_;
                int idx = (rm - rn + WS_ - 1) * (2 * WS_ - 1) + (cm - cn + WS_ - 1);
                float bb = rpb[idx * NH + head];
                float mk = amask[(size_t)(win & (NWIN - 1)) * (N_ * N_) + m * N_ + n];
                v = s[nt][r] + bb + mk;
            } else {
                v = -1e30f;
            }
            s[nt][r] = v;
        }
    }

    // ---- row softmax (row = fixed r; reduce across 16-lane half + 4 tiles) ----
#pragma unroll
    for (int r = 0; r < 8; ++r) {
        float mx = fmaxf(fmaxf(s[0][r], s[1][r]), fmaxf(s[2][r], s[3][r]));
#pragma unroll
        for (int msk = 1; msk < 16; msk <<= 1) mx = fmaxf(mx, __shfl_xor(mx, msk, 32));
        float sum = 0.f;
#pragma unroll
        for (int nt = 0; nt < 4; ++nt) {
            float e = __expf(s[nt][r] - mx);
            s[nt][r] = e;
            sum += e;
        }
#pragma unroll
        for (int msk = 1; msk < 16; msk <<= 1) sum += __shfl_xor(sum, msk, 32);
        float inv = 1.0f / sum;
#pragma unroll
        for (int nt = 0; nt < 4; ++nt) s[nt][r] *= inv;
    }

    // ---- P -> LDS (f16) ----
#pragma unroll
    for (int nt = 0; nt < 4; ++nt)
#pragma unroll
        for (int r = 0; r < 8; ++r)
            Plds[mtile + lg * 8 + r][nt * 16 + ncol] = (_Float16)s[nt][r];
    __syncthreads();

    // ---- out = P @ V : 2 d-tiles x 2 k-steps ----
#pragma unroll
    for (int dt = 0; dt < 2; ++dt) {
        v8f o = vzero8();
#pragma unroll
        for (int ks = 0; ks < 2; ++ks) {
            v16h ap;
#pragma unroll
            for (int j = 0; j < 16; ++j) {
                int vv = j >> 1;
                int kk = ks * 32 + ((vv < 4 ? 0 : 16) + lg * 8 + (vv & 3) * 2 + (j & 1));
                ap[j] = Plds[mtile + ncol][kk];
            }
            v16h bv;
#pragma unroll
            for (int j = 0; j < 16; ++j) bv[j] = Vlds[ks * 32 + (j < 8 ? 0 : 16) + lg * 8 + (j & 7)][dt * 16 + ncol];
            o = wmma16x16x32(ap, bv, o);
        }
#pragma unroll
        for (int r = 0; r < 8; ++r) {
            int m = mtile + lg * 8 + r;
            if (m < N_)
                out[(size_t)(win * N_ + m) * C_ + head * HD + dt * 16 + ncol] =
                    (_Float16)o[r];
        }
    }
}

// ---------------------------------------------------------------------------
extern "C" void kernel_launch(void* const* d_in, const int* in_sizes, int n_in,
                              void* d_out, int out_size, void* d_ws, size_t ws_size,
                              hipStream_t stream) {
    (void)in_sizes; (void)n_in; (void)out_size; (void)ws_size;
    const float* x        = (const float*)d_in[0];
    const float* amask    = (const float*)d_in[1];
    const float* g1       = (const float*)d_in[2];
    const float* b1       = (const float*)d_in[3];
    const float* qkv_w    = (const float*)d_in[4];
    const float* qkv_b    = (const float*)d_in[5];
    const float* rpb      = (const float*)d_in[6];
    const float* proj_w   = (const float*)d_in[7];
    const float* proj_b   = (const float*)d_in[8];
    const float* fc1_w    = (const float*)d_in[9];
    const float* fc1_b    = (const float*)d_in[10];
    float* out            = (float*)d_out;

    char* ws = (char*)d_ws;
    const size_t XW_OFF  = 0;                                    // 100352*192*2
    const size_t QKV_OFF = XW_OFF  + (size_t)MROWS * C_ * 2;     // 100352*576*2
    const size_t AO_OFF  = QKV_OFF + (size_t)MROWS * 576 * 2;    // 100352*192*2
    const size_t XO_OFF  = AO_OFF  + (size_t)MROWS * C_ * 2;     // 100352*192*2
    const size_t WQ_OFF  = XO_OFF  + (size_t)MROWS * C_ * 2;     // 192*576*2
    const size_t WP_OFF  = WQ_OFF  + (size_t)192 * 576 * 2;      // 192*192*2
    const size_t WF_OFF  = WP_OFF  + (size_t)192 * 192 * 2;

    _Float16* xw    = (_Float16*)(ws + XW_OFF);
    _Float16* qkvB  = (_Float16*)(ws + QKV_OFF);
    _Float16* aoB   = (_Float16*)(ws + AO_OFF);
    _Float16* xoB   = (_Float16*)(ws + XO_OFF);
    _Float16* wqT   = (_Float16*)(ws + WQ_OFF);
    _Float16* wpT   = (_Float16*)(ws + WP_OFF);
    _Float16* wfT   = (_Float16*)(ws + WF_OFF);

    // 0: weight transpose/convert (184320 elems)
    prep_weights<<<720, 256, 0, stream>>>(qkv_w, proj_w, fc1_w, wqT, wpT, wfT);

    // 1: LN + shift + partition (100352 rows, 8 rows/block)
    ln_shift_partition<<<MROWS / 8, 256, 0, stream>>>(x, g1, b1, xw);

    // 2: QKV GEMM  (M=100352, N=576)
    gemm_wmma<0><<<dim3(MROWS / 128, 576 / 64), 256, 0, stream>>>(
        xw, wqT, qkv_b, 576, nullptr, nullptr, qkvB);

    // 3: windowed attention
    attn_kernel<<<dim3(NWTOT, NH), 128, 0, stream>>>(qkvB, rpb, amask, aoB);

    // 4: PROJ GEMM + reverse shift scatter (M=100352, N=192)
    gemm_wmma<1><<<dim3(MROWS / 128, C_ / 64), 256, 0, stream>>>(
        aoB, wpT, proj_b, C_, nullptr, nullptr, xoB);

    // 5: FC1 GEMM + GELU + residual -> f32 output
    gemm_wmma<2><<<dim3(MROWS / 128, C_ / 64), 256, 0, stream>>>(
        xoB, wfT, fc1_b, C_, x, out, nullptr);
}
